// AttentionLayers_89593017795285
// MI455X (gfx1250) — matrix operand order repaired and checked by reading, not compiled, not run on hardware
//
#include <hip/hip_runtime.h>
#include <math.h>

#define B_   4
#define N_   1024
#define DIM_ 1024
#define H_   16
#define DH_  64
#define M_   16
#define JP_  1056          // M_+N_ padded up to a multiple of 32
#define HGP_ 64            // head-gate GEMM padded N (so all GEMMs have N % 64 == 0)
#define QK_SCALE 10.0f

typedef __attribute__((ext_vector_type(16))) __bf16 v16bf;
typedef __attribute__((ext_vector_type(8)))  float  v8f;
typedef __attribute__((ext_vector_type(4)))  int    v4i_;

typedef union { v16bf v; unsigned int w[8]; unsigned short u[16]; } BF16x16;

#define WMMA_BF16(A_, B_, C_) \
  __builtin_amdgcn_wmma_f32_16x16x32_bf16(false, (A_), false, (B_), (short)0, (C_), false, false)

// ---- CDNA5 async global->LDS staging (guarded: falls back to plain loads) ----
#if defined(__has_builtin)
#if __has_builtin(__builtin_amdgcn_global_load_async_to_lds_b128) && \
    __has_builtin(__builtin_amdgcn_s_wait_asynccnt)
#define HAVE_ASYNC_LDS 1
#endif
#endif

#ifdef HAVE_ASYNC_LDS
#define AS1_ __attribute__((address_space(1)))
#define AS3_ __attribute__((address_space(3)))
__device__ __forceinline__ void async_copy16(const void* g, void* l) {
  __builtin_amdgcn_global_load_async_to_lds_b128((AS1_ v4i_*)(g), (AS3_ v4i_*)(l), 0, 0);
}
#endif

__device__ __forceinline__ unsigned short f2bf(float f) {
  unsigned int u = __builtin_bit_cast(unsigned int, f);
  u += 0x7FFFu + ((u >> 16) & 1u);
  return (unsigned short)(u >> 16);
}
__device__ __forceinline__ float sigmoidf_(float x) { return 1.0f / (1.0f + __expf(-x)); }

// ---------- WMMA fragment loaders (ISA 7.12.2 layouts, wave32; 16B-aligned b128 loads) ----------
// A (16x32 bf16, row-major source): K = kb+0..7 then kb+16..23, pairs contiguous
__device__ __forceinline__ v16bf load_fragA(const unsigned short* base, int stride, int lane) {
  int row = lane & 15;
  int kb  = (lane & 16) ? 8 : 0;
  const unsigned short* p = base + (size_t)row * stride + kb;
  BF16x16 f;
  *(uint4*)&f.w[0] = *(const uint4*)(p);        // K = kb+0..7
  *(uint4*)&f.w[4] = *(const uint4*)(p + 16);   // K = kb+16..23
  return f.v;
}
// B (32x16 bf16) where source holds the TRANSPOSE row-major (K contiguous per col)
__device__ __forceinline__ v16bf load_fragB_T(const unsigned short* base, int stride, int lane) {
  int col = lane & 15;
  int kb  = (lane & 16) ? 8 : 0;
  const unsigned short* p = base + (size_t)col * stride + kb;
  BF16x16 f;
  *(uint4*)&f.w[0] = *(const uint4*)(p);        // K = kb+0..7
  *(uint4*)&f.w[4] = *(const uint4*)(p + 16);   // K = kb+16..23
  return f.v;
}

// ---------- generic fp32 -> bf16 convert ----------
__global__ void cvt_f32_bf16(const float* __restrict__ in, unsigned short* __restrict__ out, int n) {
  int i = blockIdx.x * 256 + threadIdx.x;
  if (i < n) out[i] = f2bf(in[i]);
}
// W_hgate [DIM][H] -> bf16 padded to [DIM][HGP_]
__global__ void cvt_pad_hg(const float* __restrict__ in, unsigned short* __restrict__ out) {
  int i = blockIdx.x * 256 + threadIdx.x;       // over DIM_*HGP_
  int r = i >> 6, c = i & (HGP_ - 1);
  out[i] = (c < H_) ? f2bf(in[r * H_ + c]) : (unsigned short)0;
}

// ---------- bf16 WMMA GEMM:  C[M,N] = A[M,K] * B[K,N]  (fp32 out), N % 64 == 0 ----------
// block 256 = 8 waves; workgroup tile 32(M) x 64(N); K-chunks of 32.
// A tile staged via CDNA5 async global->LDS (ASYNCcnt); B tile transposed into LDS.
__global__ __launch_bounds__(256) void gemm_bf16(const unsigned short* __restrict__ A,
                                                 const unsigned short* __restrict__ Bm,
                                                 float* __restrict__ C, int K, int N) {
  __shared__ unsigned short As[32][32];
  __shared__ unsigned short Bt[64][32];   // B tile transposed: [n][k]
  int tid = threadIdx.x;
  int lane = tid & 31, wave = tid >> 5;
  int wm = wave >> 2, wn = wave & 3;
  int m0 = blockIdx.y * 32;
  int n0 = blockIdx.x * 64;
  v8f acc = {};
  for (int k0 = 0; k0 < K; k0 += 32) {
#ifdef HAVE_ASYNC_LDS
    if (tid < 128) {                       // 128 lanes x 16B = 2KB A tile, no VGPR round-trip
      int ar = tid >> 2, ac = (tid & 3) * 8;
      async_copy16(&A[(size_t)(m0 + ar) * K + k0 + ac], &As[ar][ac]);
    }
#else
    {
      int ai = tid * 4; int ar = ai >> 5; int ac = ai & 31;
      *(uint2*)&As[ar][ac] = *(const uint2*)&A[(size_t)(m0 + ar) * K + k0 + ac];
    }
#endif
    {
      int br = tid >> 3, bc = (tid & 7) * 8;   // row k=br, cols bc..bc+7
      union { uint4 q; unsigned short s[8]; } u;
      u.q = *(const uint4*)&Bm[(size_t)(k0 + br) * N + n0 + bc];
#pragma unroll
      for (int e = 0; e < 8; ++e) Bt[bc + e][br] = u.s[e];
    }
#ifdef HAVE_ASYNC_LDS
    __builtin_amdgcn_s_wait_asynccnt(0);
#endif
    __syncthreads();
    v16bf a = load_fragA(&As[16 * wm][0], 32, lane);
    v16bf b = load_fragB_T(&Bt[16 * wn][0], 32, lane);
    acc = WMMA_BF16(a, b, acc);
    __syncthreads();
  }
  int ccol  = n0 + 16 * wn + (lane & 15);
  int crow0 = m0 + 16 * wm + ((lane & 16) ? 8 : 0);
#pragma unroll
  for (int r = 0; r < 8; ++r) C[(size_t)(crow0 + r) * N + ccol] = acc[r];
}

// ---------- l2norm + per-head scale + RoPE + relocate to head-major bf16 ----------
// mode 0: q -> dst[b,h,n,d]      (stride N_,  offset 0)
// mode 1: k -> dst[b,h,M_+n,d]   (stride JP_, offset M_)
// mode 2: v -> TRANSPOSED dst[b,h,d,M_+n]  (so attn@V B-fragments are contiguous)
__global__ void norm_rope_relocate(const float* __restrict__ raw, const float* __restrict__ freqs,
                                   const float* __restrict__ scale, unsigned short* __restrict__ dst,
                                   int mode, int seqStride, int seqOffset) {
  int idx = blockIdx.x;                  // ((b*N_ + n)*H_ + h)
  int h = idx & (H_ - 1);
  int n = (idx >> 4) & (N_ - 1);
  int b = idx >> 14;
  int d = threadIdx.x;
  float val = raw[((size_t)(b * N_ + n)) * (H_ * DH_) + h * DH_ + d];
  __shared__ float red[DH_];
  __shared__ float vv[DH_];
  if (mode < 2) {
    red[d] = val * val; __syncthreads();
    for (int s = 32; s > 0; s >>= 1) { if (d < s) red[d] += red[d + s]; __syncthreads(); }
    float nrm = sqrtf(red[0]);
    val = val / fmaxf(nrm, 1e-12f) * scale[h * DH_ + d];
    vv[d] = val; __syncthreads();
    float rot = (d & 1) ? vv[d - 1] : -vv[d + 1];
    float fr  = freqs[n * DH_ + d];
    val = val * __cosf(fr) + rot * __sinf(fr);
    dst[(((size_t)(b * H_ + h)) * seqStride + seqOffset + n) * DH_ + d] = f2bf(val);
  } else {
    dst[(((size_t)(b * H_ + h)) * DH_ + d) * JP_ + seqOffset + n] = f2bf(val);
  }
}

// ---------- memory K/V rows (positions 0..M_-1) + zero pad rows (M_+N_..JP_-1) ----------
__global__ void memkv_fill(const float* __restrict__ mem_k, const float* __restrict__ mem_v,
                           const float* __restrict__ k_scale,
                           unsigned short* __restrict__ kb, unsigned short* __restrict__ vb) {
  int idx = blockIdx.x;                  // ((b*H_ + h)*32 + r)
  int r = idx & 31;
  int h = (idx >> 5) & (H_ - 1);
  int b = idx >> 9;
  int d = threadIdx.x;
  size_t kbase = ((size_t)(b * H_ + h)) * JP_;          // k rows [j][d]
  size_t vbase = ((size_t)(b * H_ + h)) * DH_;          // v transposed [d][j]
  if (r < M_) {
    float kv = mem_k[(h * M_ + r) * DH_ + d];
    __shared__ float red[DH_];
    red[d] = kv * kv; __syncthreads();
    for (int s = 32; s > 0; s >>= 1) { if (d < s) red[d] += red[d + s]; __syncthreads(); }
    float nrm = sqrtf(red[0]);
    kb[(kbase + r) * DH_ + d] = f2bf(kv / fmaxf(nrm, 1e-12f) * k_scale[h * DH_ + d]);
    vb[(vbase + d) * JP_ + r] = f2bf(mem_v[(h * M_ + r) * DH_ + d]);
  } else {
    int row = (M_ + N_) + (r - M_);
    kb[(kbase + row) * DH_ + d] = 0;
    vb[(vbase + d) * JP_ + row] = 0;
  }
}

// ---------- attention pass 1: softmax stats (max, sum) of pre-mixed dots ----------
// block = 512 threads = 16 waves (one per head), grid = (N_/16, B_)
__global__ __launch_bounds__(512) void attn_pass1(const unsigned short* __restrict__ qb,
                                                  const unsigned short* __restrict__ kb,
                                                  const float* __restrict__ Wpre,
                                                  float* __restrict__ stats) {
  __shared__ float dotsS[H_][16][32];
  int lane = threadIdx.x & 31;
  int g    = threadIdx.x >> 5;
  int b    = blockIdx.y;
  int i0   = blockIdx.x * 16;
  int bh   = b * H_ + g;

  const unsigned short* qh = qb + (((size_t)bh) * N_ + i0) * DH_;
  v16bf qa0 = load_fragA(qh, DH_, lane);
  v16bf qa1 = load_fragA(qh + 32, DH_, lane);

  float wpre[H_];
#pragma unroll
  for (int h = 0; h < H_; ++h) wpre[h] = Wpre[g * H_ + h];

  int ccol = lane & 15;
  int crow = (lane & 16) ? 8 : 0;
  float m8[8], l8[8];
#pragma unroll
  for (int r = 0; r < 8; ++r) { m8[r] = -3.0e38f; l8[r] = 0.0f; }

  const unsigned short* kh = kb + ((size_t)bh) * JP_ * DH_;

  for (int j0 = 0; j0 < JP_; j0 += 32) {
    v8f c0 = {}; v8f c1 = {};
    {
      v16bf b0 = load_fragB_T(kh + (size_t)j0 * DH_,       DH_, lane);
      v16bf b1 = load_fragB_T(kh + (size_t)j0 * DH_ + 32,  DH_, lane);
      c0 = WMMA_BF16(qa0, b0, c0);
      c0 = WMMA_BF16(qa1, b1, c0);
      v16bf b2 = load_fragB_T(kh + (size_t)(j0 + 16) * DH_,      DH_, lane);
      v16bf b3 = load_fragB_T(kh + (size_t)(j0 + 16) * DH_ + 32, DH_, lane);
      c1 = WMMA_BF16(qa0, b2, c1);
      c1 = WMMA_BF16(qa1, b3, c1);
    }
#pragma unroll
    for (int r = 0; r < 8; ++r) {
      dotsS[g][crow + r][ccol]      = c0[r] * QK_SCALE;
      dotsS[g][crow + r][16 + ccol] = c1[r] * QK_SCALE;
    }
    __syncthreads();
    float mx0[8], mx1[8];
#pragma unroll
    for (int r = 0; r < 8; ++r) { mx0[r] = 0.0f; mx1[r] = 0.0f; }
    for (int h = 0; h < H_; ++h) {
      float w = wpre[h];
#pragma unroll
      for (int r = 0; r < 8; ++r) {
        mx0[r] += w * dotsS[h][crow + r][ccol];
        mx1[r] += w * dotsS[h][crow + r][16 + ccol];
      }
    }
    int jj0 = j0 + ccol;
    int jj1 = j0 + 16 + ccol;
#pragma unroll
    for (int r = 0; r < 8; ++r) {
      int ii = i0 + crow + r;
      if (jj0 >= M_ && (jj0 - M_) > ii) mx0[r] = -3.0e38f;
      if (jj1 >= M_ && (jj1 - M_) > ii) mx1[r] = -3.0e38f;
      float mloc = fmaxf(mx0[r], mx1[r]);
#pragma unroll
      for (int off = 1; off < 16; off <<= 1) mloc = fmaxf(mloc, __shfl_xor(mloc, off, 32));
      float mnew = fmaxf(m8[r], mloc);
      float s = __expf(mx0[r] - mnew) + __expf(mx1[r] - mnew);
#pragma unroll
      for (int off = 1; off < 16; off <<= 1) s += __shfl_xor(s, off, 32);
      l8[r] = l8[r] * __expf(m8[r] - mnew) + s;
      m8[r] = mnew;
    }
    __syncthreads();
  }
  if (ccol == 0) {
#pragma unroll
    for (int r = 0; r < 8; ++r) {
      size_t idx = ((size_t)bh) * N_ + i0 + crow + r;
      stats[2 * idx]     = m8[r];
      stats[2 * idx + 1] = l8[r];
    }
  }
}

// ---------- attention pass 2: normalized attn, post-mix, attn @ V ----------
__global__ __launch_bounds__(512) void attn_pass2(const unsigned short* __restrict__ qb,
                                                  const unsigned short* __restrict__ kb,
                                                  const unsigned short* __restrict__ vb,
                                                  const float* __restrict__ Wpre,
                                                  const float* __restrict__ Wpost,
                                                  const float* __restrict__ stats,
                                                  float* __restrict__ outp) {
  __shared__ float bufA[H_][16][32];     // dots; later reused as bf16 mixed-attn staging
  __shared__ float bufB[H_][16][32];     // normalized per-head attn
  int lane = threadIdx.x & 31;
  int g    = threadIdx.x >> 5;
  int b    = blockIdx.y;
  int i0   = blockIdx.x * 16;
  int bh   = b * H_ + g;

  const unsigned short* qh = qb + (((size_t)bh) * N_ + i0) * DH_;
  v16bf qa0 = load_fragA(qh, DH_, lane);
  v16bf qa1 = load_fragA(qh + 32, DH_, lane);

  float wpre[H_], wpost[H_];
#pragma unroll
  for (int h = 0; h < H_; ++h) { wpre[h] = Wpre[g * H_ + h]; wpost[h] = Wpost[g * H_ + h]; }

  int ccol = lane & 15;
  int crow = (lane & 16) ? 8 : 0;
  float m8[8], li8[8];
#pragma unroll
  for (int r = 0; r < 8; ++r) {
    size_t idx = ((size_t)bh) * N_ + i0 + crow + r;
    m8[r]  = stats[2 * idx];
    li8[r] = 1.0f / stats[2 * idx + 1];
  }

  const unsigned short* kh = kb + ((size_t)bh) * JP_ * DH_;
  unsigned short* ap = ((unsigned short*)&bufA[0][0][0]) + g * (16 * 32);
  v8f o[4] = {};

  for (int j0 = 0; j0 < JP_; j0 += 32) {
    v8f c0 = {}; v8f c1 = {};
    {
      v16bf b0 = load_fragB_T(kh + (size_t)j0 * DH_,       DH_, lane);
      v16bf b1 = load_fragB_T(kh + (size_t)j0 * DH_ + 32,  DH_, lane);
      c0 = WMMA_BF16(qa0, b0, c0);
      c0 = WMMA_BF16(qa1, b1, c0);
      v16bf b2 = load_fragB_T(kh + (size_t)(j0 + 16) * DH_,      DH_, lane);
      v16bf b3 = load_fragB_T(kh + (size_t)(j0 + 16) * DH_ + 32, DH_, lane);
      c1 = WMMA_BF16(qa0, b2, c1);
      c1 = WMMA_BF16(qa1, b3, c1);
    }
#pragma unroll
    for (int r = 0; r < 8; ++r) {
      bufA[g][crow + r][ccol]      = c0[r] * QK_SCALE;
      bufA[g][crow + r][16 + ccol] = c1[r] * QK_SCALE;
    }
    __syncthreads();
    // pre-softmax head mix + mask + exact softmax with precomputed stats
    {
      float mx0[8], mx1[8];
#pragma unroll
      for (int r = 0; r < 8; ++r) { mx0[r] = 0.0f; mx1[r] = 0.0f; }
      for (int h = 0; h < H_; ++h) {
        float w = wpre[h];
#pragma unroll
        for (int r = 0; r < 8; ++r) {
          mx0[r] += w * bufA[h][crow + r][ccol];
          mx1[r] += w * bufA[h][crow + r][16 + ccol];
        }
      }
      int jj0 = j0 + ccol;
      int jj1 = j0 + 16 + ccol;
#pragma unroll
      for (int r = 0; r < 8; ++r) {
        int ii = i0 + crow + r;
        if (jj0 >= M_ && (jj0 - M_) > ii) mx0[r] = -3.0e38f;
        if (jj1 >= M_ && (jj1 - M_) > ii) mx1[r] = -3.0e38f;
        bufB[g][crow + r][ccol]      = __expf(mx0[r] - m8[r]) * li8[r];
        bufB[g][crow + r][16 + ccol] = __expf(mx1[r] - m8[r]) * li8[r];
      }
    }
    __syncthreads();
    // post-softmax head mix -> bf16 staging (overlays bufA, safe: dots fully consumed)
    {
      float p0[8], p1[8];
#pragma unroll
      for (int r = 0; r < 8; ++r) { p0[r] = 0.0f; p1[r] = 0.0f; }
      for (int h = 0; h < H_; ++h) {
        float w = wpost[h];
#pragma unroll
        for (int r = 0; r < 8; ++r) {
          p0[r] += w * bufB[h][crow + r][ccol];
          p1[r] += w * bufB[h][crow + r][16 + ccol];
        }
      }
#pragma unroll
      for (int r = 0; r < 8; ++r) {
        ap[(crow + r) * 32 + ccol]      = f2bf(p0[r]);
        ap[(crow + r) * 32 + 16 + ccol] = f2bf(p1[r]);
      }
    }
    __syncthreads();
    // attnP (16x32) @ V (32x64) via 4 WMMAs; V is stored transposed [d][j]
    {
      v16bf af = load_fragA(ap, 32, lane);
      const unsigned short* vh = vb + ((size_t)bh) * DH_ * JP_ + j0;
#pragma unroll
      for (int t = 0; t < 4; ++t) {
        v16bf vf = load_fragB_T(vh + (size_t)(16 * t) * JP_, JP_, lane);
        o[t] = WMMA_BF16(af, vf, o[t]);
      }
    }
    __syncthreads();
  }
  float* orow = outp + (((size_t)bh) * N_ + i0) * DH_;
#pragma unroll
  for (int t = 0; t < 4; ++t)
#pragma unroll
    for (int r = 0; r < 8; ++r)
      orow[(crow + r) * DH_ + 16 * t + ccol] = o[t][r];
}

// ---------- gating: out * sigmoid(head_gate) * sigmoid(value_gate) -> bf16 [B*N, H*DH] ----------
__global__ void gate_pack(const float* __restrict__ outp, const float* __restrict__ hgraw,
                          const float* __restrict__ vgraw, const float* __restrict__ bhg,
                          const float* __restrict__ bvg, unsigned short* __restrict__ gated) {
  int i = blockIdx.x * 256 + threadIdx.x;         // over B*N*H*DH
  int d = i & 63;
  int h = (i >> 6) & 15;
  int n = (i >> 10) & 1023;
  int b = i >> 20;
  float val = outp[(((size_t)b * H_ + h) * N_ + n) * DH_ + d];
  float hg  = sigmoidf_(hgraw[((size_t)b * N_ + n) * HGP_ + h] + bhg[h]);
  float vg  = sigmoidf_(vgraw[((size_t)b * N_ + n) * (H_ * DH_) + h * DH_ + d] + bvg[h * DH_ + d]);
  gated[((size_t)b * N_ + n) * (H_ * DH_) + h * DH_ + d] = f2bf(val * hg * vg);
}

extern "C" void kernel_launch(void* const* d_in, const int* in_sizes, int n_in,
                              void* d_out, int out_size, void* d_ws, size_t ws_size,
                              hipStream_t stream) {
  (void)in_sizes; (void)n_in; (void)out_size; (void)ws_size;
  const float* x       = (const float*)d_in[0];
  const float* freqs   = (const float*)d_in[1];
  const float* Wq      = (const float*)d_in[2];
  const float* Wk      = (const float*)d_in[3];
  const float* Wv      = (const float*)d_in[4];
  const float* q_scale = (const float*)d_in[5];
  const float* k_scale = (const float*)d_in[6];
  const float* mem_k   = (const float*)d_in[7];
  const float* mem_v   = (const float*)d_in[8];
  const float* W_pre   = (const float*)d_in[9];
  const float* W_post  = (const float*)d_in[10];
  const float* W_hg    = (const float*)d_in[11];
  const float* b_hg    = (const float*)d_in[12];
  const float* W_vg    = (const float*)d_in[13];
  const float* b_vg    = (const float*)d_in[14];
  const float* Wo      = (const float*)d_in[15];
  float* out = (float*)d_out;

  char* ws = (char*)d_ws;
  size_t off = 0;
  auto alloc = [&](size_t bytes) -> char* {
    char* p = ws + off; off += (bytes + 255) & ~(size_t)255; return p;
  };
  unsigned short* xb   = (unsigned short*)alloc((size_t)B_ * N_ * DIM_ * 2);
  unsigned short* wqb  = (unsigned short*)alloc((size_t)DIM_ * H_ * DH_ * 2);
  unsigned short* wkb  = (unsigned short*)alloc((size_t)DIM_ * H_ * DH_ * 2);
  unsigned short* wvb  = (unsigned short*)alloc((size_t)DIM_ * H_ * DH_ * 2);
  unsigned short* wvgb = (unsigned short*)alloc((size_t)DIM_ * H_ * DH_ * 2);
  unsigned short* wob  = (unsigned short*)alloc((size_t)H_ * DH_ * DIM_ * 2);
  unsigned short* whgb = (unsigned short*)alloc((size_t)DIM_ * HGP_ * 2);
  float*          raw  = (float*)alloc((size_t)B_ * N_ * H_ * DH_ * 4);
  unsigned short* qbf  = (unsigned short*)alloc((size_t)B_ * H_ * N_ * DH_ * 2);
  unsigned short* kbf  = (unsigned short*)alloc((size_t)B_ * H_ * JP_ * DH_ * 2);
  unsigned short* vbf  = (unsigned short*)alloc((size_t)B_ * H_ * DH_ * JP_ * 2);
  float*          stats= (float*)alloc((size_t)B_ * H_ * N_ * 2 * 4);
  float*          outp = (float*)alloc((size_t)B_ * H_ * N_ * DH_ * 4);
  float*          hgraw= (float*)alloc((size_t)B_ * N_ * HGP_ * 4);
  unsigned short* gated= (unsigned short*)alloc((size_t)B_ * N_ * DIM_ * 2);
  float*          vgraw= raw;   // reused after V is relocated

  const int BN = B_ * N_;
  // 1) bf16 conversions
  cvt_f32_bf16<<<(BN * DIM_ + 255) / 256, 256, 0, stream>>>(x,    xb,   BN * DIM_);
  cvt_f32_bf16<<<(DIM_ * DIM_ + 255) / 256, 256, 0, stream>>>(Wq,  wqb,  DIM_ * DIM_);
  cvt_f32_bf16<<<(DIM_ * DIM_ + 255) / 256, 256, 0, stream>>>(Wk,  wkb,  DIM_ * DIM_);
  cvt_f32_bf16<<<(DIM_ * DIM_ + 255) / 256, 256, 0, stream>>>(Wv,  wvb,  DIM_ * DIM_);
  cvt_f32_bf16<<<(DIM_ * DIM_ + 255) / 256, 256, 0, stream>>>(W_vg, wvgb, DIM_ * DIM_);
  cvt_f32_bf16<<<(DIM_ * DIM_ + 255) / 256, 256, 0, stream>>>(Wo,  wob,  DIM_ * DIM_);
  cvt_pad_hg<<<(DIM_ * HGP_ + 255) / 256, 256, 0, stream>>>(W_hg, whgb);

  // 2) projections + norm/rope/relocate (raw buffer reused between them)
  dim3 gg(DIM_ / 64, BN / 32);
  gemm_bf16<<<gg, 256, 0, stream>>>(xb, wqb, raw, DIM_, H_ * DH_);
  norm_rope_relocate<<<BN * H_, 64, 0, stream>>>(raw, freqs, q_scale, qbf, 0, N_, 0);
  gemm_bf16<<<gg, 256, 0, stream>>>(xb, wkb, raw, DIM_, H_ * DH_);
  norm_rope_relocate<<<BN * H_, 64, 0, stream>>>(raw, freqs, k_scale, kbf, 1, JP_, M_);
  gemm_bf16<<<gg, 256, 0, stream>>>(xb, wvb, raw, DIM_, H_ * DH_);
  norm_rope_relocate<<<BN * H_, 64, 0, stream>>>(raw, freqs, k_scale, vbf, 2, JP_, M_);
  memkv_fill<<<B_ * H_ * 32, 64, 0, stream>>>(mem_k, mem_v, k_scale, kbf, vbf);

  // 3) gate projections (value-gate GEMM reuses `raw`)
  dim3 ghg(1, BN / 32);
  gemm_bf16<<<ghg, 256, 0, stream>>>(xb, whgb, hgraw, DIM_, HGP_);
  gemm_bf16<<<gg, 256, 0, stream>>>(xb, wvgb, vgraw, DIM_, H_ * DH_);

  // 4) two-pass attention with talking heads
  dim3 ga(N_ / 16, B_);
  attn_pass1<<<ga, 512, 0, stream>>>(qbf, kbf, W_pre, stats);
  attn_pass2<<<ga, 512, 0, stream>>>(qbf, kbf, vbf, W_pre, W_post, stats, outp);

  // 5) gating + output projection
  gate_pack<<<(B_ * N_ * H_ * DH_ + 255) / 256, 256, 0, stream>>>(outp, hgraw, vgraw, b_hg, b_vg, gated);
  gemm_bf16<<<gg, 256, 0, stream>>>(gated, wob, out, H_ * DH_, DIM_);
}
